// WindowAttention2d_41738492182656
// MI455X (gfx1250) — compile-verified, matrix operand order checked by reading, not run on hardware
//
#include <hip/hip_runtime.h>
#include <hip/hip_bf16.h>

typedef __attribute__((ext_vector_type(16))) _Float16 v16h;
typedef __attribute__((ext_vector_type(8)))  _Float16 v8h;
typedef __attribute__((ext_vector_type(8)))  float    v8f;

#define WIN    8
#define SHIFT  4
#define HEADS  3
#define C_DIM  96
#define L_TOK  64
#define HRES   256

// LDS layout (bytes), phases overlaid:
//  [0,     24576): xbufT f32[96][64] (channel-major) -> sbuf f32[64][64] (16384) + pbuf f16[64][64] @16384
//  [24576, 36864): ybuf f16[64][96]   ->  obuf f16[64][96]
//  [36864, 61440): qkbuf f16[64][192] ->  t1buf f16[64][96]
//  [61440, 73728): vtbuf f16[96][64]
#define SMEM_BYTES 73728

// Load a 16x32 f16 WMMA A/B fragment from a row-major [rows][ld] f16 matrix.
// ISA layout: lanes 0-15 row=lane, halves0-7 = K k0..k0+7, halves8-15 = K k0+16..k0+23;
// lanes 16-31 shift kb by +8. (B operand identical with lane indexing the N column,
// i.e. matrix stored as B^T row-major.)
__device__ __forceinline__ v16h load_frag(const _Float16* base, int ld, int row0, int k0, int lane) {
  int row = row0 + (lane & 15);
  int kb  = k0 + ((lane >> 4) << 3);
  const _Float16* p = base + row * ld + kb;
  v8h lo = *reinterpret_cast<const v8h*>(p);
  v8h hi = *reinterpret_cast<const v8h*>(p + 16);
  return __builtin_shufflevector(lo, hi, 0,1,2,3,4,5,6,7,8,9,10,11,12,13,14,15);
}

__global__ void cvt_weights_kernel(const float* __restrict__ wq,
                                   const float* __restrict__ wo,
                                   const float* __restrict__ wp,
                                   _Float16* __restrict__ dst) {
  const int NQ = 3 * C_DIM * C_DIM;   // 288x96
  const int NO = C_DIM * C_DIM;       // 96x96
  const int total = NQ + 2 * NO;
  for (int i = blockIdx.x * blockDim.x + threadIdx.x; i < total; i += gridDim.x * blockDim.x) {
    float v;
    if (i < NQ)            v = wq[i];
    else if (i < NQ + NO)  v = wo[i - NQ];
    else                   v = wp[i - NQ - NO];
    dst[i] = (_Float16)v;
  }
}

__global__ void __launch_bounds__(128)
win_attn_kernel(const float* __restrict__ x,
                const float* __restrict__ ln_g, const float* __restrict__ ln_b,
                const float* __restrict__ b_qkv, const float* __restrict__ b_out,
                const float* __restrict__ b_proj,
                const _Float16* __restrict__ wq16,
                const _Float16* __restrict__ wout16,
                const _Float16* __restrict__ wproj16,
                float* __restrict__ out)
{
  extern __shared__ char smem[];
  float*    xbufT = reinterpret_cast<float*>(smem);               // [96][64] channel-major
  float*    sbuf  = reinterpret_cast<float*>(smem);               // [64][64]
  _Float16* pbuf  = reinterpret_cast<_Float16*>(smem + 16384);    // [64][64]
  _Float16* ybuf  = reinterpret_cast<_Float16*>(smem + 24576);    // [64][96]
  _Float16* obuf  = ybuf;                                         // reuse after QKV
  _Float16* qkbuf = reinterpret_cast<_Float16*>(smem + 36864);    // [64][192] (q|k)
  _Float16* t1buf = qkbuf;                                        // reuse after attention
  _Float16* vtbuf = reinterpret_cast<_Float16*>(smem + 61440);    // [96][64] = V^T

  const int tid  = threadIdx.x;
  const int lane = tid & 31;
  const int wave = tid >> 5;

  const int wid = blockIdx.x;           // 8192 windows
  const int b   = wid >> 10;
  const int wi  = (wid >> 5) & 31;
  const int wj  = wid & 31;

  // keep f16 weights hot in cache (global_prefetch_b8)
  for (int offb = tid * 128; offb < (3*C_DIM*C_DIM + 2*C_DIM*C_DIM) * 2; offb += 128 * 128)
    __builtin_prefetch((const char*)wq16 + offb, 0, 3);

  // ---- phase 0: async gather of the shifted window straight into LDS ----
  // Because SHIFT=4 is a multiple of the 4-float chunk size, every 16B chunk of a
  // window row stays contiguous in memory even when the roll wraps at 256.
  // 96 channels x 8 rows x 2 chunks = 1536 b128 transfers (12 per thread).
  for (int idx = tid; idx < C_DIM * 16; idx += 128) {
    int c    = idx >> 4;
    int rem  = idx & 15;
    int i    = rem >> 1;        // token row 0..7
    int half = rem & 1;         // which 4-float chunk of the row
    int t0   = i * 8 + half * 4;
    int gh   = (wi * WIN + i + SHIFT) & (HRES - 1);
    int gw0  = (wj * WIN + half * 4 + SHIFT) & (HRES - 1);   // chunk-aligned, 4 contiguous
    const float* gp = x + (((size_t)b * C_DIM + c) * HRES + gh) * HRES + gw0;
    unsigned int lds_off = (unsigned int)(uintptr_t)(xbufT + (c * L_TOK + t0));
    unsigned long long gaddr = (unsigned long long)(uintptr_t)gp;
    asm volatile("global_load_async_to_lds_b128 %0, %1, off"
                 :: "v"(lds_off), "v"(gaddr) : "memory");
  }
  asm volatile("s_wait_asynccnt 0x0" ::: "memory");
  __syncthreads();

  // ---- phase 1: LayerNorm per token -> ybuf f16 (xbufT is channel-major) ----
  if (tid < L_TOK) {
    float s = 0.f, ss = 0.f;
    for (int c = 0; c < C_DIM; ++c) { float v = xbufT[c * L_TOK + tid]; s += v; ss += v * v; }
    float mu  = s * (1.f / C_DIM);
    float var = ss * (1.f / C_DIM) - mu * mu;
    float rs  = rsqrtf(var + 1e-5f);
    for (int c = 0; c < C_DIM; ++c) {
      float v = (xbufT[c * L_TOK + tid] - mu) * rs * ln_g[c] + ln_b[c];
      ybuf[tid * C_DIM + c] = (_Float16)v;
    }
  }
  __syncthreads();

  // ---- phase 2: QKV GEMM  D[64x288] = ybuf[64x96] * w_qkv^T  (+ b_qkv) ----
  // q,k -> qkbuf[64][192]; v -> vtbuf transposed [96][64]
  for (int tile = wave; tile < 4 * 18; tile += 4) {
    int mt = tile & 3, nt = tile >> 2;
    int m0 = mt * 16, n0 = nt * 16;
    int n  = n0 + (lane & 15);
    v8f acc;
    float bias = b_qkv[n];
    for (int r = 0; r < 8; ++r) acc[r] = bias;
    for (int k0 = 0; k0 < C_DIM; k0 += 32) {
      v16h a  = load_frag(ybuf, C_DIM, m0, k0, lane);
      v16h bb = load_frag(wq16, C_DIM, n0, k0, lane);
      acc = __builtin_amdgcn_wmma_f32_16x16x32_f16(false, a, false, bb, (short)0, acc, false, false);
    }
    int mb = m0 + ((lane >> 4) << 3);
    if (n0 < 192) {
      for (int r = 0; r < 8; ++r) qkbuf[(mb + r) * 192 + n] = (_Float16)acc[r];
    } else {
      for (int r = 0; r < 8; ++r) vtbuf[(n - 192) * 64 + mb + r] = (_Float16)acc[r];
    }
  }
  __syncthreads();

  const float scale = 0.17677669529663687f;   // 1/sqrt(32)

  // ---- phase 3: per-head attention ----
  for (int h = 0; h < HEADS; ++h) {
    // S = Q_h * K_h^T  (M=64,N=64,K=32: one WMMA per tile)
    for (int tile = wave; tile < 16; tile += 4) {
      int mt = tile & 3, nt = tile >> 2;
      int m0 = mt * 16, n0 = nt * 16;
      v8f acc;
      for (int r = 0; r < 8; ++r) acc[r] = 0.f;
      v16h a  = load_frag(qkbuf, 192, m0, h * 32, lane);
      v16h bb = load_frag(qkbuf, 192, n0, 96 + h * 32, lane);
      acc = __builtin_amdgcn_wmma_f32_16x16x32_f16(false, a, false, bb, (short)0, acc, false, false);
      int n  = n0 + (lane & 15);
      int mb = m0 + ((lane >> 4) << 3);
      for (int r = 0; r < 8; ++r) sbuf[(mb + r) * 64 + n] = acc[r] * scale;
    }
    __syncthreads();

    // softmax rows -> pbuf f16
    if (tid < L_TOK) {
      const float* row = sbuf + tid * 64;
      float mx = row[0];
      for (int n = 1; n < 64; ++n) mx = fmaxf(mx, row[n]);
      float sum = 0.f;
      for (int n = 0; n < 64; ++n) sum += __expf(row[n] - mx);
      float inv = 1.f / sum;
      for (int n = 0; n < 64; ++n) pbuf[tid * 64 + n] = (_Float16)(__expf(row[n] - mx) * inv);
    }
    __syncthreads();

    // O_h = P * V_h  (M=64,N=32,K=64)
    for (int tile = wave; tile < 8; tile += 4) {
      int mt = tile & 3, nt = tile >> 2;
      int m0 = mt * 16, n0 = nt * 16;
      v8f acc;
      for (int r = 0; r < 8; ++r) acc[r] = 0.f;
      for (int k0 = 0; k0 < 64; k0 += 32) {
        v16h a  = load_frag(pbuf, 64, m0, k0, lane);
        v16h bb = load_frag(vtbuf, 64, h * 32 + n0, k0, lane);
        acc = __builtin_amdgcn_wmma_f32_16x16x32_f16(false, a, false, bb, (short)0, acc, false, false);
      }
      int n  = n0 + (lane & 15);
      int mb = m0 + ((lane >> 4) << 3);
      for (int r = 0; r < 8; ++r) obuf[(mb + r) * 96 + h * 32 + n] = (_Float16)acc[r];
    }
    __syncthreads();
  }

  // ---- phase 4: out projection  t1 = obuf * w_out^T + b_out ----
  for (int tile = wave; tile < 24; tile += 4) {
    int mt = tile & 3, nt = tile >> 2;
    int m0 = mt * 16, n0 = nt * 16;
    int n  = n0 + (lane & 15);
    v8f acc;
    float bias = b_out[n];
    for (int r = 0; r < 8; ++r) acc[r] = bias;
    for (int k0 = 0; k0 < C_DIM; k0 += 32) {
      v16h a  = load_frag(obuf, C_DIM, m0, k0, lane);
      v16h bb = load_frag(wout16, C_DIM, n0, k0, lane);
      acc = __builtin_amdgcn_wmma_f32_16x16x32_f16(false, a, false, bb, (short)0, acc, false, false);
    }
    int mb = m0 + ((lane >> 4) << 3);
    for (int r = 0; r < 8; ++r) t1buf[(mb + r) * 96 + n] = (_Float16)acc[r];
  }
  __syncthreads();

  // ---- phase 5: final projection + rolled scatter to global ----
  for (int tile = wave; tile < 24; tile += 4) {
    int mt = tile & 3, nt = tile >> 2;
    int m0 = mt * 16, n0 = nt * 16;
    int n  = n0 + (lane & 15);
    v8f acc;
    float bias = b_proj[n];
    for (int r = 0; r < 8; ++r) acc[r] = bias;
    for (int k0 = 0; k0 < C_DIM; k0 += 32) {
      v16h a  = load_frag(t1buf, C_DIM, m0, k0, lane);
      v16h bb = load_frag(wproj16, C_DIM, n0, k0, lane);
      acc = __builtin_amdgcn_wmma_f32_16x16x32_f16(false, a, false, bb, (short)0, acc, false, false);
    }
    int mb = m0 + ((lane >> 4) << 3);
    for (int r = 0; r < 8; ++r) {
      int t = mb + r;
      int i = t >> 3, j = t & 7;
      int gh = (wi * WIN + i + SHIFT) & (HRES - 1);
      int gw = (wj * WIN + j + SHIFT) & (HRES - 1);
      out[(((size_t)b * C_DIM + n) * HRES + gh) * HRES + gw] = acc[r];
    }
  }
}

extern "C" void kernel_launch(void* const* d_in, const int* in_sizes, int n_in,
                              void* d_out, int out_size, void* d_ws, size_t ws_size,
                              hipStream_t stream) {
  const float* x      = (const float*)d_in[0];
  const float* ln_g   = (const float*)d_in[1];
  const float* ln_b   = (const float*)d_in[2];
  const float* w_qkv  = (const float*)d_in[3];
  const float* b_qkv  = (const float*)d_in[4];
  const float* w_out  = (const float*)d_in[5];
  const float* b_out  = (const float*)d_in[6];
  const float* w_proj = (const float*)d_in[7];
  const float* b_proj = (const float*)d_in[8];
  float* out = (float*)d_out;

  _Float16* w16 = (_Float16*)d_ws;
  cvt_weights_kernel<<<90, 256, 0, stream>>>(w_qkv, w_out, w_proj, w16);

  const _Float16* wq16    = w16;
  const _Float16* wout16  = w16 + 3 * C_DIM * C_DIM;
  const _Float16* wproj16 = wout16 + C_DIM * C_DIM;

  win_attn_kernel<<<dim3(8192), dim3(128), SMEM_BYTES, stream>>>(
      x, ln_g, ln_b, b_qkv, b_out, b_proj, wq16, wout16, wproj16, out);
}
